// MultiHeadAttention_33303176413738
// MI455X (gfx1250) — matrix operand order repaired and checked by reading, not compiled, not run on hardware
//
#include <hip/hip_runtime.h>

// ---------------------------------------------------------------------------
// MI455X (gfx1250) multi-head attention, bf16 WMMA + double-buffered async
// global->LDS staging (ASYNCcnt pipeline):
//   cvt(X) / cvt+transpose(W) -> GEMM(Q,K,V) -> flash-attn -> GEMM(out)+bias
// Shapes: B=2, T=2048, D=2048, H=16, HD=128.
// K is produced in [B,H,HD,T] layout so every LDS staging step is a pure
// contiguous copy done with global_load_async_to_lds_b128; tile i+1 is DMA'd
// while tile i is consumed by v_wmma (async loads complete in order, so a
// partial s_wait_asynccnt releases the current tile only).
// ---------------------------------------------------------------------------

typedef unsigned short u16;
typedef unsigned int   u32;
typedef unsigned long long u64;
typedef __attribute__((ext_vector_type(16))) __bf16 v16bf;
typedef __attribute__((ext_vector_type(8)))  float  v8f;
typedef __attribute__((ext_vector_type(4)))  u32    v4u;
typedef __attribute__((ext_vector_type(4)))  float  v4f;
typedef __attribute__((ext_vector_type(4)))  u16    v4h;

#define NEGF (-3.0e38f)

static __device__ __forceinline__ u16 f2bf(float f) {
  u32 x = __builtin_bit_cast(u32, f);
  x += 0x7FFFu + ((x >> 16) & 1u);          // round-to-nearest-even
  return (u16)(x >> 16);
}

static __device__ __forceinline__ v16bf make_frag(v4u lo, v4u hi) {
  union { v4u q[2]; v16bf v; } u;
  u.q[0] = lo; u.q[1] = hi;
  return u.v;
}

// LDS byte offset of a generic pointer to __shared__ (LDS aperture = low 32b).
static __device__ __forceinline__ u32 lds_off(const void* p) {
  return (u32)(size_t)p;
}

// Async global->LDS 16-byte copy (per active lane). Tracked by ASYNCcnt.
static __device__ __forceinline__ void async_b128(u32 lds, const void* gaddr) {
  asm volatile("global_load_async_to_lds_b128 %0, %1, off"
               :: "v"(lds), "v"((u64)(size_t)gaddr) : "memory");
}
template <int N>
static __device__ __forceinline__ void wait_async() {
  asm volatile("s_wait_asynccnt %0" :: "i"(N) : "memory");
}

// ---------------------------------------------------------------------------
// fp32 -> bf16 conversion (n multiple of 4)
// ---------------------------------------------------------------------------
__global__ __launch_bounds__(256)
void mha_cvt_bf16(const float* __restrict__ in, u16* __restrict__ out, int n) {
  int i = (blockIdx.x * 256 + threadIdx.x) * 4;
  if (i + 3 < n) {
    v4f v = *(const v4f*)(in + i);
    v4h o;
    o.x = f2bf(v.x); o.y = f2bf(v.y); o.z = f2bf(v.z); o.w = f2bf(v.w);
    *(v4h*)(out + i) = o;
  }
}

// ---------------------------------------------------------------------------
// W[N,K] fp32 -> Wt[K,N] bf16 (64x64 tiles via LDS)
// ---------------------------------------------------------------------------
__global__ __launch_bounds__(256)
void mha_cvt_transpose(const float* __restrict__ W, u16* __restrict__ Wt,
                       int N, int K) {
  __shared__ u16 tile[64][72] __attribute__((aligned(16)));
  const int tid = threadIdx.x;
  const int k0 = blockIdx.x * 64, n0 = blockIdx.y * 64;
  {
    int n = tid >> 2, kq = (tid & 3) * 16;
    const float* src = W + (size_t)(n0 + n) * K + k0 + kq;
#pragma unroll
    for (int j = 0; j < 16; j += 4) {
      v4f v = *(const v4f*)(src + j);
      tile[n][kq + j + 0] = f2bf(v.x);
      tile[n][kq + j + 1] = f2bf(v.y);
      tile[n][kq + j + 2] = f2bf(v.z);
      tile[n][kq + j + 3] = f2bf(v.w);
    }
  }
  __syncthreads();
  {
    int k = tid >> 2, nq = (tid & 3) * 16;
    u16* dst = Wt + (size_t)(k0 + k) * N + n0 + nq;
#pragma unroll
    for (int j = 0; j < 16; j++) dst[j] = tile[nq + j][k];
  }
}

// ---------------------------------------------------------------------------
// C[M,N] = A[M,K] * Bt[K,N]   (bf16 in, f32 accumulate)
// Block 128x128, K-step 32; 8 waves -> 64x32 regions (4x2 WMMA tiles each).
// Double-buffered async staging: tile i+1 DMA overlaps tile i WMMAs.
// kt=1  : write C in K^T head layout [B,H,HD,T] (bf16)
// Cb    : bf16 out; else Cf f32 out (+bias)
// ---------------------------------------------------------------------------
#define BM 128
#define BN 128
#define BK 32

__global__ __launch_bounds__(256)
void mha_gemm_bf16(const u16* __restrict__ A, const u16* __restrict__ Bt,
                   u16* __restrict__ Cb, float* __restrict__ Cf,
                   const float* __restrict__ bias, int M, int N, int K,
                   int kt) {
  __shared__ u16 As[2][BM][BK + 8] __attribute__((aligned(16)));   // [m][k]
  __shared__ u16 Bs[2][BK][BN + 8] __attribute__((aligned(16)));   // [k][n]

  const int tid  = threadIdx.x;
  const int wave = tid >> 5, lane = tid & 31;
  const int wm   = wave >> 2, wn = wave & 3;       // 2x4 wave grid
  const int M0   = blockIdx.y * BM, N0 = blockIdx.x * BN;
  const int ln   = lane & 15, lh = lane >> 4;

  // staging coordinates (pure contiguous copies)
  const int ar = tid >> 1, ah = tid & 1;            // A: row, 16-col half
  const int bk = tid >> 3, bo = tid & 7;            // B: k-row, 16-col oct
  const u16* agp = A  + (size_t)(M0 + ar) * K + ah * 16;
  const u16* bgp = Bt + (size_t)bk * N + N0 + bo * 16;
  u32 alds[2], blds[2];
#pragma unroll
  for (int q = 0; q < 2; q++) {
    alds[q] = lds_off(&As[q][ar][ah * 16]);
    blds[q] = lds_off(&Bs[q][bk][bo * 16]);
  }

  auto issue = [&](int buf, int kk) {               // 4 async b128 per wave
    const u16* ap = agp + kk;
    const u16* bp = bgp + (size_t)kk * N;
    async_b128(alds[buf],      ap);
    async_b128(alds[buf] + 16, ap + 8);
    async_b128(blds[buf],      bp);
    async_b128(blds[buf] + 16, bp + 8);
  };

  v8f acc[4][2];
#pragma unroll
  for (int i = 0; i < 4; i++)
#pragma unroll
    for (int j = 0; j < 2; j++) acc[i][j] = v8f{};

  issue(0, 0);                                      // prologue: tile 0
  int cur = 0;
  for (int k0 = 0; k0 < K; k0 += BK) {
    if (k0 + BK < K) {
      issue(cur ^ 1, k0 + BK);                      // DMA next tile
      wait_async<4>();                              // release current tile only
    } else {
      wait_async<0>();
    }
    if (k0 + 2 * BK < K) {                          // L2 prefetch of tile i+2
      __builtin_prefetch(agp + k0 + 2 * BK, 0, 3);
      __builtin_prefetch(bgp + (size_t)(k0 + 2 * BK) * N, 0, 3);
    }
    __syncthreads();

    v16bf af[4], bfg[2];
#pragma unroll
    for (int mt = 0; mt < 4; mt++) {                // A frag: lane=m, elems=k
      const u16* row = &As[cur][wm * 64 + mt * 16 + ln][0];
      af[mt] = make_frag(*(const v4u*)(row + lh * 8),
                         *(const v4u*)(row + 16 + lh * 8));
    }
#pragma unroll
    for (int nt = 0; nt < 2; nt++) {                // B frag: lane=n, elems=k
      const u16* row = &Bs[cur][0][wn * 32 + nt * 16 + ln];
#pragma unroll
      for (int s = 0; s < 16; s++)
        bfg[nt][s] = __builtin_bit_cast(__bf16, row[((s & 7) + lh * 8 + (s >> 3) * 16) * (BN + 8)]);
    }
#pragma unroll
    for (int mt = 0; mt < 4; mt++)
#pragma unroll
      for (int nt = 0; nt < 2; nt++)
        acc[mt][nt] = __builtin_amdgcn_wmma_f32_16x16x32_bf16(
            false, af[mt], false, bfg[nt], (short)0, acc[mt][nt], false, false);
    __syncthreads();                                // buffer reuse fence
    cur ^= 1;
  }

  // epilogue: C layout -> lane holds n=ln, rows m = lh*8 + r
#pragma unroll
  for (int mt = 0; mt < 4; mt++)
#pragma unroll
    for (int nt = 0; nt < 2; nt++) {
      int gn = N0 + wn * 32 + nt * 16 + ln;
      float bv = bias ? bias[gn] : 0.f;
#pragma unroll
      for (int r = 0; r < 8; r++) {
        int gm  = M0 + wm * 64 + mt * 16 + lh * 8 + r;
        float v = acc[mt][nt][r];
        if (kt) {          // K^T layout: [B,H,HD,T], T=2048, H=16, HD=128
          int bq = gm >> 11, t = gm & 2047;
          int h  = gn >> 7,  d = gn & 127;
          Cb[((((size_t)bq * 16 + h) * 128 + d) << 11) + t] = f2bf(v);
        } else if (Cb) {
          Cb[(size_t)gm * N + gn] = f2bf(v);
        } else {
          Cf[(size_t)gm * N + gn] = v + bv;
        }
      }
    }
}

// ---------------------------------------------------------------------------
// Causal flash attention. Workgroup: one (b,h) x 128 q-rows; wave: 16 q-rows.
// Q,V in [B,T,D]; Kt in [B,H,HD,T]; O (bf16) in [B,T,D].
// Double-buffered async staging of 64-row K/V blocks.
// ---------------------------------------------------------------------------
#define HD 128

__global__ __launch_bounds__(256)
void mha_flash_attn(const u16* __restrict__ Q, const u16* __restrict__ Kt,
                    const u16* __restrict__ V, u16* __restrict__ O,
                    int T, int H) {
  __shared__ u16 Kst[2][HD][72]    __attribute__((aligned(16))); // K^T: [d][kv]
  __shared__ u16 Vs[2][64][HD + 8] __attribute__((aligned(16))); // V:   [kv][d]
  __shared__ u16 Pls[8][16][32]    __attribute__((aligned(16))); // per-wave P

  const int tid  = threadIdx.x, wave = tid >> 5, lane = tid & 31;
  const int ln   = lane & 15, lh = lane >> 4;
  const int bh   = blockIdx.y, b = bh / H, hh = bh % H;
  const int D    = H * HD;
  const int q0   = blockIdx.x * 128;
  const int wq0  = q0 + wave * 16;
  const float scale = 0.08838834764831845f;     // 1/sqrt(128)

  // staging coordinates (pure contiguous copies, 4 x b128/thread each)
  const int kd = tid >> 1, khf = tid & 1;       // K: dim row, 32-kv half
  const int vr = tid >> 2, vq = tid & 3;        // V: kv row, 32-dim quarter
  const u16* kgp = Kt + ((size_t)(b * H + hh) * HD + kd) * T + khf * 32;
  const u16* vgp = V + ((size_t)(b * T + vr)) * D + hh * HD + vq * 32;
  u32 klds[2], vlds[2];
#pragma unroll
  for (int q = 0; q < 2; q++) {
    klds[q] = lds_off(&Kst[q][kd][khf * 32]);
    vlds[q] = lds_off(&Vs[q][vr][vq * 32]);
  }

  auto stage = [&](int buf, int kvb) {          // 8 async b128 per wave
#pragma unroll
    for (int j = 0; j < 4; j++)
      async_b128(klds[buf] + j * 16, kgp + kvb + j * 8);
#pragma unroll
    for (int j = 0; j < 4; j++)
      async_b128(vlds[buf] + j * 16, vgp + (size_t)kvb * D + j * 8);
  };

  // resident Q fragments (A-layout): 16 rows x 128 dims = 4 frags of K=32
  v16bf qf[4];
  {
    const u16* qrow = Q + ((size_t)(b * T + wq0 + ln)) * D + hh * HD;
#pragma unroll
    for (int kc = 0; kc < 4; kc++)
      qf[kc] = make_frag(*(const v4u*)(qrow + kc * 32 + lh * 8),
                         *(const v4u*)(qrow + kc * 32 + 16 + lh * 8));
  }

  v8f acc[8];
#pragma unroll
  for (int i = 0; i < 8; i++) acc[i] = v8f{};
  float mi[8], li[8];
#pragma unroll
  for (int r = 0; r < 8; r++) { mi[r] = NEGF; li[r] = 0.f; }

  const int kv_end = q0 + 128;
  stage(0, 0);                                  // prologue: block 0
  int cur = 0;
  for (int kvb = 0; kvb < kv_end; kvb += 64) {
    if (kvb + 64 < kv_end) {
      stage(cur ^ 1, kvb + 64);                 // DMA next K/V block
      wait_async<8>();                          // release current block only
    } else {
      wait_async<0>();
    }
    __syncthreads();

#pragma unroll
    for (int c32 = 0; c32 < 2; c32++) {
      int kv32 = kvb + c32 * 32;
      if (kv32 <= wq0 + 15) {
        // --- S = Q K^T for 32 kv cols (two 16x16 tiles) -------------------
        v8f s[2];
#pragma unroll
        for (int st = 0; st < 2; st++) {
          int kv16 = kv32 + st * 16;
          if (kv16 <= wq0 + 15) {
            v8f sv = v8f{};
#pragma unroll
            for (int kc = 0; kc < 4; kc++) {
              const u16* col = &Kst[cur][kc * 32][c32 * 32 + st * 16 + ln];
              v16bf kf;
#pragma unroll
              for (int s = 0; s < 16; s++)
                kf[s] = __builtin_bit_cast(__bf16, col[((s & 7) + lh * 8 + (s >> 3) * 16) * 72]);
              sv = __builtin_amdgcn_wmma_f32_16x16x32_bf16(
                  false, qf[kc], false,
                  kf,
                  (short)0, sv, false, false);
            }
#pragma unroll
            for (int r = 0; r < 8; r++) {
              int row = wq0 + lh * 8 + r;
              int cc  = kv16 + ln;
              s[st][r] = (cc <= row) ? sv[r] * scale : NEGF;
            }
          } else {
#pragma unroll
            for (int r = 0; r < 8; r++) s[st][r] = NEGF;
          }
        }
        // --- online softmax (row reductions across 16 lanes of a half) ----
        float pa[8], pb[8], alpha[8];
#pragma unroll
        for (int r = 0; r < 8; r++) {
          float mx = fmaxf(s[0][r], s[1][r]);
          for (int o = 8; o >= 1; o >>= 1) mx = fmaxf(mx, __shfl_xor(mx, o, 32));
          float nm = fmaxf(mi[r], mx);
          float a  = __expf(mi[r] - nm);
          pa[r] = __expf(s[0][r] - nm);
          pb[r] = __expf(s[1][r] - nm);
          float rs = pa[r] + pb[r];
          for (int o = 8; o >= 1; o >>= 1) rs += __shfl_xor(rs, o, 32);
          li[r]    = li[r] * a + rs;
          mi[r]    = nm;
          alpha[r] = a;
        }
#pragma unroll
        for (int nt = 0; nt < 8; nt++)
#pragma unroll
          for (int r = 0; r < 8; r++) acc[nt][r] = acc[nt][r] * alpha[r];
        // --- relayout P (C-layout) -> A-layout via per-wave LDS -----------
#pragma unroll
        for (int r = 0; r < 8; r++) {
          Pls[wave][lh * 8 + r][ln]      = f2bf(pa[r]);
          Pls[wave][lh * 8 + r][16 + ln] = f2bf(pb[r]);
        }
        const u16* prow = &Pls[wave][ln][0];
        v16bf pf = make_frag(*(const v4u*)(prow + lh * 8),
                             *(const v4u*)(prow + 16 + lh * 8));
        // --- O += P V (8 dim-tiles) ---------------------------------------
#pragma unroll
        for (int nt = 0; nt < 8; nt++) {
          const u16* vrow = &Vs[cur][c32 * 32][nt * 16 + ln];
          v16bf vf;
#pragma unroll
          for (int s = 0; s < 16; s++)
            vf[s] = __builtin_bit_cast(__bf16, vrow[((s & 7) + lh * 8 + (s >> 3) * 16) * (HD + 8)]);
          acc[nt] = __builtin_amdgcn_wmma_f32_16x16x32_bf16(
              false, pf, false,
              vf,
              (short)0, acc[nt], false, false);
        }
      }
    }
    __syncthreads();                            // buffer reuse fence
    cur ^= 1;
  }

  // epilogue: normalize by row-sum, store bf16 into [B,T,D]
#pragma unroll
  for (int r = 0; r < 8; r++) {
    float rl = 1.f / li[r];
    size_t rowoff = ((size_t)(b * T + wq0 + lh * 8 + r)) * D + hh * HD;
#pragma unroll
    for (int nt = 0; nt < 8; nt++)
      O[rowoff + nt * 16 + ln] = f2bf(acc[nt][r] * rl);
  }
}

// ---------------------------------------------------------------------------
// Host launcher
// ---------------------------------------------------------------------------
extern "C" void kernel_launch(void* const* d_in, const int* in_sizes, int n_in,
                              void* d_out, int out_size, void* d_ws, size_t ws_size,
                              hipStream_t stream) {
  (void)in_sizes; (void)n_in; (void)out_size; (void)ws_size;
  const int Bsz = 2, T = 2048, Dm = 2048, H = 16;
  const size_t MT = (size_t)Bsz * T;          // 4096 rows

  const float* x  = (const float*)d_in[0];
  // d_in[1] = attention_mask (exact causal tril; implemented analytically)
  const float* Wq = (const float*)d_in[2];
  const float* Wk = (const float*)d_in[3];
  const float* Wv = (const float*)d_in[4];
  const float* Wo = (const float*)d_in[5];
  const float* bo = (const float*)d_in[6];

  u16* ws = (u16*)d_ws;
  size_t off = 0;
  u16* Xb  = ws + off; off += MT * Dm;
  u16* Wqt = ws + off; off += (size_t)Dm * Dm;
  u16* Wkt = ws + off; off += (size_t)Dm * Dm;
  u16* Wvt = ws + off; off += (size_t)Dm * Dm;
  u16* Wot = ws + off; off += (size_t)Dm * Dm;
  u16* Qb  = ws + off; off += MT * Dm;
  u16* Kb  = ws + off; off += MT * Dm;      // [B,H,HD,T] layout
  u16* Vb  = ws + off; off += MT * Dm;
  u16* AOb = ws + off; off += MT * Dm;

  mha_cvt_bf16<<<dim3((unsigned)((MT * Dm / 4 + 255) / 256)), 256, 0, stream>>>(
      x, Xb, (int)(MT * Dm));

  dim3 gt(Dm / 64, Dm / 64);                  // (32, 32)
  mha_cvt_transpose<<<gt, 256, 0, stream>>>(Wq, Wqt, Dm, Dm);
  mha_cvt_transpose<<<gt, 256, 0, stream>>>(Wk, Wkt, Dm, Dm);
  mha_cvt_transpose<<<gt, 256, 0, stream>>>(Wv, Wvt, Dm, Dm);
  mha_cvt_transpose<<<gt, 256, 0, stream>>>(Wo, Wot, Dm, Dm);

  dim3 gg(Dm / BN, (unsigned)(MT / BM));      // (16, 32)
  mha_gemm_bf16<<<gg, 256, 0, stream>>>(Xb, Wqt, Qb, nullptr, nullptr,
                                        (int)MT, Dm, Dm, 0);
  mha_gemm_bf16<<<gg, 256, 0, stream>>>(Xb, Wkt, Kb, nullptr, nullptr,
                                        (int)MT, Dm, Dm, 1);
  mha_gemm_bf16<<<gg, 256, 0, stream>>>(Xb, Wvt, Vb, nullptr, nullptr,
                                        (int)MT, Dm, Dm, 0);

  dim3 ga(T / 128, Bsz * H);                  // (16, 32)
  mha_flash_attn<<<ga, 256, 0, stream>>>(Qb, Kb, Vb, AOb, T, H);

  mha_gemm_bf16<<<gg, 256, 0, stream>>>(AOb, Wot, nullptr, (float*)d_out, bo,
                                        (int)MT, Dm, Dm, 0);
}
